// PointsEncoder_pointwise_51367808860750
// MI455X (gfx1250) — matrix operand order repaired and checked by reading, not compiled, not run on hardware
//
#include <hip/hip_runtime.h>

// ---------------------------------------------------------------------------
// Types for CDNA5 WMMA / TDM
// ---------------------------------------------------------------------------
typedef __attribute__((ext_vector_type(16))) _Float16 v16h;
typedef __attribute__((ext_vector_type(8)))  _Float16 v8h;
typedef __attribute__((ext_vector_type(8)))  float    v8f;
typedef __attribute__((ext_vector_type(4)))  unsigned int u32x4;
typedef __attribute__((ext_vector_type(8)))  int      i32x8;
typedef __attribute__((ext_vector_type(4)))  int      i32x4;

#if __has_builtin(__builtin_amdgcn_tensor_load_to_lds)
#define HAVE_TDM 1
#else
#define HAVE_TDM 0
#endif

// ---------------------------------------------------------------------------
// Weight prep: f32 (co x ci) -> f16 (co x ciPad), ciPad = roundup(ci,32).
// ---------------------------------------------------------------------------
__global__ void prep_w(const float* __restrict__ W, _Float16* __restrict__ Wh,
                       int co, int ci, int ciPad)
{
    int t = blockIdx.x * blockDim.x + threadIdx.x;
    if (t >= co * ciPad) return;
    int r = t / ciPad, c = t % ciPad;
    Wh[t] = (c < ci) ? (_Float16)W[(size_t)r * ci + c] : (_Float16)0.0f;
}

// Pack epilogue params into one [3][co] f32 tensor: bias / scale / shift.
__global__ void prep_pack(const float* __restrict__ b, const float* __restrict__ s,
                          const float* __restrict__ t, float* __restrict__ P, int co)
{
    int i = blockIdx.x * blockDim.x + threadIdx.x;
    if (i >= co) return;
    P[i]          = b[i];
    P[co + i]     = s ? s[i] : 1.0f;
    P[2 * co + i] = t ? t[i] : 0.0f;
}

// ---------------------------------------------------------------------------
// Fused conv GEMM:  Y[co,ncol] = act( (W[co,ci] @ X[ci,ncol] + b)*s + t )
// Block = 256 threads = 8 waves.  Block tile 64(co) x 128(col).
// Wave (waveCo = w>>1, waveCol = w&1) computes a 16(co) x 64(col) strip:
// one A fragment (direct f16 global b128 loads) x 4 B fragments (LDS) ->
// 4 v_wmma_f32_16x16x32_f16 per K step.  Double-buffered LDS hides VMEM
// latency behind WMMA.  Epilogue params arrive via one TENSOR_LOAD_TO_LDS
// DMA (2D tile 64x3 of the packed [3][co] tensor) overlapped with the
// whole K loop; drained with s_wait_tensorcnt.
// ---------------------------------------------------------------------------
#define TILE_CO  64
#define TILE_COL 128
#define TILE_K   32
#define LDSX_PAD 8   // row stride (32+8)*2 = 80B, 16B aligned

__global__ __launch_bounds__(256)
void conv_gemm_wmma(const _Float16* __restrict__ Wh, const float* __restrict__ X,
                    const float* __restrict__ P, float* __restrict__ Y,
                    int co, int ci, int ciPad, int ncol, int relu, int addOut)
{
    __shared__ __align__(16) _Float16 ldsX[2][TILE_COL][TILE_K + LDSX_PAD]; // [buf][col][k]
    __shared__ __align__(16) float    ldsP[3 * TILE_CO];                    // bias|scale|shift

    const int tid     = threadIdx.x;
    const int lane    = tid & 31;
    const int wave    = tid >> 5;        // 0..7
    const int waveCo  = wave >> 1;       // 0..3  (co sub-tile)
    const int waveCol = wave & 1;        // 0..1  (64-col half)

    const int coBase  = blockIdx.y * TILE_CO;
    const int colBase = blockIdx.x * TILE_COL;

    const int m = lane & 15;
    const int h = lane >> 4;

    v8f acc0 = {}, acc1 = {}, acc2 = {}, acc3 = {};

    int aRow = coBase + waveCo * 16 + m;
    if (aRow >= co) aRow = co - 1;                 // keep A loads unconditional
    const _Float16* arowBase = Wh + (size_t)aRow * ciPad;

    // ---- epilogue params: one TDM DMA of a 64x3 tile from P[3][co] ----
#if HAVE_TDM
    if (wave == 0) {
        unsigned ldsOff = (unsigned)(size_t)(&ldsP[0]);
        unsigned long long ga = (unsigned long long)(size_t)(P + coBase);
        u32x4 g0 = {};
        i32x8 g1 = {};
        i32x4 g2 = {}, g3 = {};
        i32x8 g4 = {};
        g0[0] = 1u;                                            // count=1, user D#
        g0[1] = ldsOff;                                        // lds_addr
        g0[2] = (unsigned)(ga & 0xFFFFFFFFu);                  // global_addr lo
        g0[3] = (unsigned)((ga >> 32) & 0x01FFFFFFu) | (2u << 30); // addr hi | type=2
        g1[0] = (2 << 16);                                     // data_size = 4B
        g1[1] = (int)((unsigned)(co & 0xFFFF) << 16);          // tensor_dim0 lo
        g1[2] = (int)(((co >> 16) & 0xFFFF) | (3u << 16));     // dim0 hi | tensor_dim1=3
        g1[3] = (int)(64u << 16);                              // tile_dim0 = 64
        g1[4] = 3;                                             // tile_dim1 = 3
        g1[5] = co;                                            // tensor_dim0_stride
        g1[6] = 0; g1[7] = 0;
        __builtin_amdgcn_tensor_load_to_lds(g0, g1, g2, g3, g4, 0);
    }
#else
    for (int i = tid; i < 3 * TILE_CO; i += 256) {
        int r = i >> 6, c = i & 63;
        int g = coBase + c;
        ldsP[i] = (g < co) ? P[(size_t)r * co + g] : 0.0f;
    }
#endif

    // staging geometry: thread owns columns [c4, c4+4) at k rows kkB + {0,8,16,24}
    const int kkB = tid >> 5;            // 0..7
    const int c4  = (tid & 31) * 4;      // 0,4,...,124

    // ---- stage one K tile into ldsX[buf] ----
    auto stageTile = [&](int buf, int k0) {
        const bool fast = (k0 + TILE_K <= ci) && (colBase + TILE_COL <= ncol);
        if (fast) {
            const float* colp = X + colBase + c4;
            float4 r0 = *(const float4*)(colp + (size_t)(k0 + kkB +  0) * ncol);
            float4 r1 = *(const float4*)(colp + (size_t)(k0 + kkB +  8) * ncol);
            float4 r2 = *(const float4*)(colp + (size_t)(k0 + kkB + 16) * ncol);
            float4 r3 = *(const float4*)(colp + (size_t)(k0 + kkB + 24) * ncol);
            ldsX[buf][c4 + 0][kkB +  0] = (_Float16)r0.x;
            ldsX[buf][c4 + 1][kkB +  0] = (_Float16)r0.y;
            ldsX[buf][c4 + 2][kkB +  0] = (_Float16)r0.z;
            ldsX[buf][c4 + 3][kkB +  0] = (_Float16)r0.w;
            ldsX[buf][c4 + 0][kkB +  8] = (_Float16)r1.x;
            ldsX[buf][c4 + 1][kkB +  8] = (_Float16)r1.y;
            ldsX[buf][c4 + 2][kkB +  8] = (_Float16)r1.z;
            ldsX[buf][c4 + 3][kkB +  8] = (_Float16)r1.w;
            ldsX[buf][c4 + 0][kkB + 16] = (_Float16)r2.x;
            ldsX[buf][c4 + 1][kkB + 16] = (_Float16)r2.y;
            ldsX[buf][c4 + 2][kkB + 16] = (_Float16)r2.z;
            ldsX[buf][c4 + 3][kkB + 16] = (_Float16)r2.w;
            ldsX[buf][c4 + 0][kkB + 24] = (_Float16)r3.x;
            ldsX[buf][c4 + 1][kkB + 24] = (_Float16)r3.y;
            ldsX[buf][c4 + 2][kkB + 24] = (_Float16)r3.z;
            ldsX[buf][c4 + 3][kkB + 24] = (_Float16)r3.w;
        } else {
            #pragma unroll
            for (int u = 0; u < 4; ++u) {
                int kk = kkB + u * 8;
                int gk = k0 + kk;
                int gc = colBase + c4;
                float4 v = make_float4(0.0f, 0.0f, 0.0f, 0.0f);
                if (gk < ci && gc < ncol)
                    v = *(const float4*)(X + (size_t)gk * ncol + gc);
                ldsX[buf][c4 + 0][kk] = (_Float16)v.x;
                ldsX[buf][c4 + 1][kk] = (_Float16)v.y;
                ldsX[buf][c4 + 2][kk] = (_Float16)v.z;
                ldsX[buf][c4 + 3][kk] = (_Float16)v.w;
            }
        }
    };

    // ---- prologue: stage tile 0, load A fragment 0 ----
    stageTile(0, 0);
    v8h alo = *(const v8h*)(arowBase + h * 16);
    v8h ahi = *(const v8h*)(arowBase + h * 16 + 8);
    __syncthreads();

    const int nTiles = ciPad / TILE_K;
    for (int t = 0; t < nTiles; ++t) {
        const int buf = t & 1;
        const int k0  = t * TILE_K;

        // ---- issue next tile's staging + next A fragment before computing ----
        v8h nalo = alo, nahi = ahi;
        const bool more = (t + 1 < nTiles);
        if (more) {
            stageTile(buf ^ 1, k0 + TILE_K);
            nalo = *(const v8h*)(arowBase + (k0 + TILE_K) + h * 16);
            nahi = *(const v8h*)(arowBase + (k0 + TILE_K) + h * 16 + 8);
            if (k0 + 2 * TILE_K < ci)   // prefetch tile t+2 (global_prefetch_b8)
                __builtin_prefetch(&X[(size_t)(k0 + 2 * TILE_K) * ncol + colBase + c4], 0, 1);
        }

        v16h afrag = __builtin_shufflevector(alo, ahi,
                                             0, 1, 2, 3, 4, 5, 6, 7,
                                             8, 9, 10, 11, 12, 13, 14, 15);

        // ---- 4 B fragments from ldsX[buf], 4 WMMAs ----
        {
            const _Float16* brow = &ldsX[buf][waveCol * 64 + 0 * 16 + m][0];
            v8h blo = *(const v8h*)(brow + h * 16);
            v8h bhi = *(const v8h*)(brow + h * 16 + 8);
            v16h bfrag = __builtin_shufflevector(blo, bhi,
                                                 0, 1, 2, 3, 4, 5, 6, 7,
                                                 8, 9, 10, 11, 12, 13, 14, 15);
            acc0 = __builtin_amdgcn_wmma_f32_16x16x32_f16(false, afrag, false, bfrag,
                                                          (short)0, acc0, false, false);
        }
        {
            const _Float16* brow = &ldsX[buf][waveCol * 64 + 1 * 16 + m][0];
            v8h blo = *(const v8h*)(brow + h * 16);
            v8h bhi = *(const v8h*)(brow + h * 16 + 8);
            v16h bfrag = __builtin_shufflevector(blo, bhi,
                                                 0, 1, 2, 3, 4, 5, 6, 7,
                                                 8, 9, 10, 11, 12, 13, 14, 15);
            acc1 = __builtin_amdgcn_wmma_f32_16x16x32_f16(false, afrag, false, bfrag,
                                                          (short)0, acc1, false, false);
        }
        {
            const _Float16* brow = &ldsX[buf][waveCol * 64 + 2 * 16 + m][0];
            v8h blo = *(const v8h*)(brow + h * 16);
            v8h bhi = *(const v8h*)(brow + h * 16 + 8);
            v16h bfrag = __builtin_shufflevector(blo, bhi,
                                                 0, 1, 2, 3, 4, 5, 6, 7,
                                                 8, 9, 10, 11, 12, 13, 14, 15);
            acc2 = __builtin_amdgcn_wmma_f32_16x16x32_f16(false, afrag, false, bfrag,
                                                          (short)0, acc2, false, false);
        }
        {
            const _Float16* brow = &ldsX[buf][waveCol * 64 + 3 * 16 + m][0];
            v8h blo = *(const v8h*)(brow + h * 16);
            v8h bhi = *(const v8h*)(brow + h * 16 + 8);
            v16h bfrag = __builtin_shufflevector(blo, bhi,
                                                 0, 1, 2, 3, 4, 5, 6, 7,
                                                 8, 9, 10, 11, 12, 13, 14, 15);
            acc3 = __builtin_amdgcn_wmma_f32_16x16x32_f16(false, afrag, false, bfrag,
                                                          (short)0, acc3, false, false);
        }

        alo = nalo; ahi = nahi;
        __syncthreads();
    }

    // ---- drain the TDM DMA, make params visible to all waves ----
#if HAVE_TDM
    if (wave == 0) __builtin_amdgcn_s_wait_tensorcnt(0);
    __syncthreads();
#endif

    // ---- epilogue: C/D layout lane L: n = L&15, VGPR j -> m = 8*(L>>4)+j ----
    const int n  = lane & 15;
    const int mh = (lane >> 4) * 8;
    v8f* accs[4] = { &acc0, &acc1, &acc2, &acc3 };
    #pragma unroll
    for (int cc = 0; cc < 4; ++cc) {
        int col = colBase + waveCol * 64 + cc * 16 + n;
        if (col >= ncol) continue;
        v8f a = *accs[cc];
        #pragma unroll
        for (int j = 0; j < 8; ++j) {
            int localCo = waveCo * 16 + mh + j;
            int gco = coBase + localCo;
            if (gco < co) {
                float v = (a[j] + ldsP[localCo]) * ldsP[64 + localCo] + ldsP[128 + localCo];
                if (relu)  v = v > 0.0f ? v : 0.0f;
                size_t o = (size_t)gco * ncol + col;
                if (addOut) Y[o] += v; else Y[o] = v;
            }
        }
    }
}

// ---------------------------------------------------------------------------
// Farthest point sampling: one block per batch, iterative argmax of running
// min-distance kept in LDS, block-wide reduction.
// ---------------------------------------------------------------------------
__global__ __launch_bounds__(256)
void fps_kernel(const float* __restrict__ xyz, int n, int npoint, int* __restrict__ out)
{
    __shared__ float dmin[8192];
    __shared__ float redv[256];
    __shared__ int   redi[256];
    __shared__ int   far_s;

    const float* xs = xyz;
    const float* ys = xyz + n;
    const float* zs = xyz + 2 * n;

    for (int i = threadIdx.x; i < n; i += 256) dmin[i] = 1e10f;
    if (threadIdx.x == 0) far_s = 0;
    __syncthreads();

    for (int it = 0; it < npoint; ++it) {
        int far = far_s;
        if (threadIdx.x == 0) out[it] = far;
        float fx = xs[far], fy = ys[far], fz = zs[far];
        float best = -1.0f; int bi = 0;
        for (int i = threadIdx.x; i < n; i += 256) {
            float dx = xs[i] - fx, dy = ys[i] - fy, dz = zs[i] - fz;
            float d  = dx * dx + dy * dy + dz * dz;
            float dm = fminf(dmin[i], d);
            dmin[i] = dm;
            if (dm > best) { best = dm; bi = i; }
        }
        redv[threadIdx.x] = best; redi[threadIdx.x] = bi;
        __syncthreads();
        for (int s = 128; s > 0; s >>= 1) {
            if (threadIdx.x < s && redv[threadIdx.x + s] > redv[threadIdx.x]) {
                redv[threadIdx.x] = redv[threadIdx.x + s];
                redi[threadIdx.x] = redi[threadIdx.x + s];
            }
            __syncthreads();
        }
        if (threadIdx.x == 0) far_s = redi[0];
        __syncthreads();
    }
}

__global__ void gather_rows3(const float* __restrict__ src, int n,
                             const int* __restrict__ idx, int m, float* __restrict__ dst)
{
    int t = blockIdx.x * blockDim.x + threadIdx.x;
    if (t >= m) return;
    int id = idx[t];
    dst[t]         = src[id];
    dst[m + t]     = src[n + id];
    dst[2 * m + t] = src[2 * n + id];
}

__global__ void ball_query_kernel(const float* __restrict__ xyz, int n,
                                  const float* __restrict__ nq, int npnt,
                                  int ns, float r2, int* __restrict__ out)
{
    int q = blockIdx.x * blockDim.x + threadIdx.x;
    if (q >= npnt) return;
    float qx = nq[q], qy = nq[npnt + q], qz = nq[2 * npnt + q];
    int cnt = 0, first = 0;
    for (int i = 0; i < n && cnt < ns; ++i) {
        float dx = xyz[i] - qx, dy = xyz[n + i] - qy, dz = xyz[2 * n + i] - qz;
        if (dx * dx + dy * dy + dz * dz < r2) {
            if (cnt == 0) first = i;
            out[q * ns + cnt] = i;
            ++cnt;
        }
    }
    for (; cnt < ns; ++cnt) out[q * ns + cnt] = first;
}

// grouped = concat([xyz[idx]-new_xyz, feats[idx]]) : (3+cf, npnt, ns)
__global__ void group_sa_kernel(const float* __restrict__ xyz, int nx,
                                const float* __restrict__ feats, int cf,
                                const float* __restrict__ newxyz, int npnt,
                                const int* __restrict__ idx, int ns,
                                float* __restrict__ out)
{
    int t = blockIdx.x * blockDim.x + threadIdx.x;
    if (t >= npnt * ns) return;
    int q = t / ns;
    int id = idx[t];
    size_t ncol = (size_t)npnt * ns;
    for (int c = 0; c < 3; ++c)
        out[c * ncol + t] = xyz[(size_t)c * nx + id] - newxyz[(size_t)c * npnt + q];
    for (int c = 0; c < cf; ++c)
        out[(size_t)(3 + c) * ncol + t] = feats[(size_t)c * nx + id];
}

__global__ void maxpool_ns(const float* __restrict__ in, int cnp, int ns,
                           float* __restrict__ out)
{
    int t = blockIdx.x * blockDim.x + threadIdx.x;
    if (t >= cnp) return;
    const float* p = in + (size_t)t * ns;
    float m = p[0];
    for (int s = 1; s < ns; ++s) m = fmaxf(m, p[s]);
    out[t] = m;
}

// 16-NN (includes self), register insertion sort
__global__ void knn_kernel(const float* __restrict__ pos, int np, int* __restrict__ idx)
{
    int q = blockIdx.x * blockDim.x + threadIdx.x;
    if (q >= np) return;
    float qx = pos[q], qy = pos[np + q], qz = pos[2 * np + q];
    float bd[16]; int bi[16];
    #pragma unroll
    for (int j = 0; j < 16; ++j) { bd[j] = 3.4e38f; bi[j] = 0; }
    for (int i = 0; i < np; ++i) {
        float dx = pos[i] - qx, dy = pos[np + i] - qy, dz = pos[2 * np + i] - qz;
        float d = dx * dx + dy * dy + dz * dz;
        if (d < bd[15]) {
            bd[15] = d; bi[15] = i;
            #pragma unroll
            for (int t = 15; t > 0; --t) {
                if (bd[t] < bd[t - 1]) {
                    float td = bd[t]; bd[t] = bd[t - 1]; bd[t - 1] = td;
                    int ti = bi[t]; bi[t] = bi[t - 1]; bi[t - 1] = ti;
                }
            }
        }
    }
    #pragma unroll
    for (int j = 0; j < 16; ++j) idx[q * 16 + j] = bi[j];
}

__global__ void group_feat(const float* __restrict__ f, int c, int np,
                           const int* __restrict__ idx, int K, float* __restrict__ out)
{
    int t = blockIdx.x * blockDim.x + threadIdx.x;
    if (t >= np * K) return;
    int id = idx[t];
    size_t ncol = (size_t)np * K;
    for (int cc = 0; cc < c; ++cc)
        out[cc * ncol + t] = f[(size_t)cc * np + id];
}

__global__ void pos_rel_kernel(const float* __restrict__ pos, int np,
                               const int* __restrict__ idx, int K, float* __restrict__ out)
{
    int t = blockIdx.x * blockDim.x + threadIdx.x;
    if (t >= np * K) return;
    int q = t / K, id = idx[t];
    size_t ncol = (size_t)np * K;
    for (int c = 0; c < 3; ++c)
        out[c * ncol + t] = pos[(size_t)c * np + q] - pos[(size_t)c * np + id];
}

// attn_in = q[...,None] - k_grouped + pe
__global__ void qkpe_kernel(const float* __restrict__ Q, const float* __restrict__ Kg,
                            const float* __restrict__ PE, int c, int np, int K,
                            float* __restrict__ out)
{
    size_t total = (size_t)c * np * K;
    size_t t = (size_t)blockIdx.x * blockDim.x + threadIdx.x;
    if (t >= total) return;
    size_t ncol = (size_t)np * K;
    int cc = (int)(t / ncol);
    int col = (int)(t % ncol);
    int q = col / K;
    out[t] = Q[(size_t)cc * np + q] - Kg[t] + PE[t];
}

__global__ void softmax_k(float* __restrict__ a, int cnp, int K)
{
    int t = blockIdx.x * blockDim.x + threadIdx.x;
    if (t >= cnp) return;
    float* p = a + (size_t)t * K;
    float m = p[0];
    for (int j = 1; j < K; ++j) m = fmaxf(m, p[j]);
    float s = 0.0f;
    for (int j = 0; j < K; ++j) { float e = __expf(p[j] - m); p[j] = e; s += e; }
    float inv = 1.0f / s;
    for (int j = 0; j < K; ++j) p[j] *= inv;
}

// agg[c,q] = sum_j attn[c,q,j] * (v[c,q] + pe[c,q,j])
__global__ void aggregate_kernel(const float* __restrict__ attn, const float* __restrict__ V,
                                 const float* __restrict__ PE, int cnp, int K,
                                 float* __restrict__ agg)
{
    int t = blockIdx.x * blockDim.x + threadIdx.x;
    if (t >= cnp) return;
    float v = V[t];
    const float* at = attn + (size_t)t * K;
    const float* pe = PE + (size_t)t * K;
    float s = 0.0f;
    for (int j = 0; j < K; ++j) s += at[j] * (v + pe[j]);
    agg[t] = s;
}

// three-NN inverse-distance interpolation, fused
__global__ void three_interp(const float* __restrict__ x1, int n1,
                             const float* __restrict__ x2, int n2,
                             const float* __restrict__ p2, int c2,
                             float* __restrict__ out)
{
    int i = blockIdx.x * blockDim.x + threadIdx.x;
    if (i >= n1) return;
    float qx = x1[i], qy = x1[n1 + i], qz = x1[2 * n1 + i];
    float d0 = 3.4e38f, d1 = 3.4e38f, d2 = 3.4e38f;
    int i0 = 0, i1 = 0, i2 = 0;
    for (int j = 0; j < n2; ++j) {
        float dx = x2[j] - qx, dy = x2[n2 + j] - qy, dz = x2[2 * n2 + j] - qz;
        float d = dx * dx + dy * dy + dz * dz;
        if (d < d0)      { d2 = d1; i2 = i1; d1 = d0; i1 = i0; d0 = d; i0 = j; }
        else if (d < d1) { d2 = d1; i2 = i1; d1 = d; i1 = j; }
        else if (d < d2) { d2 = d; i2 = j; }
    }
    if (n2 < 2) { d1 = d0; i1 = i0; }
    if (n2 < 3) { d2 = d1; i2 = i1; }
    float w0 = 1.0f / fmaxf(sqrtf(fmaxf(d0, 0.0f)), 1e-10f);
    float w1 = 1.0f / fmaxf(sqrtf(fmaxf(d1, 0.0f)), 1e-10f);
    float w2 = 1.0f / fmaxf(sqrtf(fmaxf(d2, 0.0f)), 1e-10f);
    float ws = w0 + w1 + w2;
    w0 /= ws; w1 /= ws; w2 /= ws;
    for (int c = 0; c < c2; ++c) {
        const float* pr = p2 + (size_t)c * n2;
        out[(size_t)c * n1 + i] = w0 * pr[i0] + w1 * pr[i1] + w2 * pr[i2];
    }
}

__global__ void copy_flat(const float* __restrict__ src, float* __restrict__ dst, size_t n)
{
    size_t t = (size_t)blockIdx.x * blockDim.x + threadIdx.x;
    if (t < n) dst[t] = src[t];
}

__global__ void add_inplace(float* __restrict__ y, const float* __restrict__ x, size_t n)
{
    size_t t = (size_t)blockIdx.x * blockDim.x + threadIdx.x;
    if (t < n) y[t] += x[t];
}

__global__ void fill_kernel(float* __restrict__ p, size_t n, float v)
{
    size_t t = (size_t)blockIdx.x * blockDim.x + threadIdx.x;
    if (t < n) p[t] = v;
}

// ---------------------------------------------------------------------------
// Host orchestration
// ---------------------------------------------------------------------------
struct ConvP {
    const float* W; const float* b; const float* s; const float* t;
    const _Float16* Wh; const float* pars; int ci, ciPad, co;
};
struct TransP { ConvP attn1, attn2, end, key, pos1, pos2, query, start, value; };

#define G1(n) dim3((unsigned)(((n) + 255) / 256)), dim3(256), 0, st

static inline void run_conv(hipStream_t st, const ConvP& p, const float* X, float* Y,
                            int ncol, bool relu, bool addOut = false)
{
    dim3 grid((ncol + TILE_COL - 1) / TILE_COL, (p.co + TILE_CO - 1) / TILE_CO);
    conv_gemm_wmma<<<grid, 256, 0, st>>>(p.Wh, X, p.pars, Y,
                                         p.co, p.ci, p.ciPad, ncol,
                                         relu ? 1 : 0, addOut ? 1 : 0);
}

static void run_transformer(hipStream_t st, const TransP& tp, float* x, const float* pos,
                            int ci, int np, int* knnidx,
                            float* Xs, float* Q, float* Kt, float* V, float* Prel,
                            float* Kg, float* PE, float* T0, float* T1)
{
    const int K = 16;
    size_t nk = (size_t)np * K;
    knn_kernel<<<G1(np)>>>(pos, np, knnidx);
    run_conv(st, tp.start, x, Xs, np, false);
    run_conv(st, tp.query, Xs, Q, np, false);
    run_conv(st, tp.key, Xs, Kt, np, false);
    run_conv(st, tp.value, Xs, V, np, false);
    group_feat<<<G1(nk)>>>(Kt, 64, np, knnidx, K, Kg);
    pos_rel_kernel<<<G1(nk)>>>(pos, np, knnidx, K, Prel);
    run_conv(st, tp.pos1, Prel, T0, (int)nk, true);
    run_conv(st, tp.pos2, T0, PE, (int)nk, false);
    qkpe_kernel<<<G1(64 * nk)>>>(Q, Kg, PE, 64, np, K, T0);
    run_conv(st, tp.attn1, T0, T1, (int)nk, true);
    run_conv(st, tp.attn2, T1, T0, (int)nk, false);
    softmax_k<<<G1(64 * np)>>>(T0, 64 * np, K);
    aggregate_kernel<<<G1(64 * np)>>>(T0, V, PE, 64 * np, K, Q);   // agg -> Q
    run_conv(st, tp.end, Q, x, np, false, /*addOut=*/true);        // x += end(agg)
}

extern "C" void kernel_launch(void* const* d_in, const int* in_sizes, int n_in,
                              void* d_out, int out_size, void* d_ws, size_t ws_size,
                              hipStream_t stream)
{
    (void)in_sizes; (void)n_in; (void)out_size; (void)ws_size;
    hipStream_t st = stream;
    const int Bv = 4, Nv = 8192;

    // ---- workspace bump allocator ----
    char* ws = (char*)d_ws;
    size_t off = 0;
    auto alloc = [&](size_t bytes) -> void* {
        void* p = ws + off; off += (bytes + 255) & ~(size_t)255; return p;
    };

    // ---- params: jax pytree flatten order (dicts sorted by key) ----
    int cur = 1; // d_in[0] = point_cloud
    auto nextConv = [&](int ci, int co, bool bn) -> ConvP {
        ConvP p;
        p.W = (const float*)d_in[cur++];
        p.b = (const float*)d_in[cur++];
        p.s = bn ? (const float*)d_in[cur++] : nullptr;
        p.t = bn ? (const float*)d_in[cur++] : nullptr;
        p.ci = ci; p.co = co;
        p.ciPad = (ci + 31) & ~31;
        _Float16* wh = (_Float16*)alloc((size_t)co * p.ciPad * sizeof(_Float16));
        prep_w<<<G1(co * p.ciPad)>>>(p.W, wh, co, ci, p.ciPad);
        p.Wh = wh;
        float* pars = (float*)alloc((size_t)3 * co * sizeof(float));
        prep_pack<<<G1(co)>>>(p.b, p.s, p.t, pars, co);
        p.pars = pars;
        return p;
    };
    auto nextTrans = [&](int ci) -> TransP {
        TransP t;                                  // sorted keys:
        t.attn1 = nextConv(64, 256, true);         // attn1
        t.attn2 = nextConv(256, 64, false);        // attn2
        t.end   = nextConv(64, ci, false);         // end
        t.key   = nextConv(64, 64, false);         // key
        t.pos1  = nextConv(3, 64, true);           // pos1
        t.pos2  = nextConv(64, 64, false);         // pos2
        t.query = nextConv(64, 64, false);         // query
        t.start = nextConv(ci, 64, false);         // start
        t.value = nextConv(64, 64, false);         // value
        return t;
    };

    ConvP down1   = nextConv(128, 128, true);
    ConvP down2   = nextConv(256, 256, true);
    ConvP fp1c[3] = { nextConv(134, 128, true), nextConv(128, 128, true), nextConv(128, 128, true) };
    ConvP fp2c[2] = { nextConv(384, 256, true), nextConv(256, 128, true) };
    ConvP fp3c[2] = { nextConv(768, 256, true), nextConv(256, 256, true) };
    ConvP sa1c[3] = { nextConv(10, 64, true), nextConv(64, 64, true), nextConv(64, 128, true) };
    ConvP sa2c[3] = { nextConv(131, 128, true), nextConv(128, 128, true), nextConv(128, 256, true) };
    ConvP sa3c[3] = { nextConv(259, 256, true), nextConv(256, 512, true), nextConv(512, 512, true) };
    TransP t1 = nextTrans(128), t2 = nextTrans(256), te1 = nextTrans(128), te2 = nextTrans(256);

    // ---- activation / index buffers ----
    float* l1xyz = (float*)alloc((size_t)Bv * 3 * 1024 * 4);
    float* l1    = (float*)alloc((size_t)Bv * 128 * 1024 * 4);
    float* l2xyz = (float*)alloc((size_t)Bv * 3 * 256 * 4);
    float* l2    = (float*)alloc((size_t)Bv * 256 * 256 * 4);
    float* l2up  = (float*)alloc((size_t)Bv * 256 * 256 * 4);
    float* l1up  = (float*)alloc((size_t)Bv * 128 * 1024 * 4);
    int*   fpsidx  = (int*)alloc(1024 * 4);
    int*   ballidx = (int*)alloc(1024 * 64 * 4);
    int*   knnidx  = (int*)alloc(1024 * 16 * 4);
    float* zero3   = (float*)alloc(256);
    float* S0 = (float*)alloc((size_t)40 << 20);
    float* S1 = (float*)alloc((size_t)40 << 20);
    float* S2 = (float*)alloc((size_t)40 << 20);
    float* S3 = (float*)alloc((size_t)20 << 20);
    float* S4 = (float*)alloc((size_t)20 << 20);

    fill_kernel<<<G1(4)>>>(zero3, 4, 0.0f);

    // transformer scratch carving (np <= 1024, K = 16)
    float* Xs   = S3;
    float* Q    = S3 + 1 * 65536;
    float* Kt   = S3 + 2 * 65536;
    float* V    = S3 + 3 * 65536;
    float* Prel = S3 + 4 * 65536;
    float* Kg   = S4;
    float* PE   = S4 + 1048576;
    float* T0   = S4 + 2 * 1048576;

    float* outp = (float*)d_out;

    for (int b = 0; b < Bv; ++b) {
        const float* pc = (const float*)d_in[0] + (size_t)b * 7 * Nv;  // (7, N)
        float* l1xyz_b = l1xyz + (size_t)b * 3 * 1024;
        float* l1_b    = l1    + (size_t)b * 128 * 1024;
        float* l2xyz_b = l2xyz + (size_t)b * 3 * 256;
        float* l2_b    = l2    + (size_t)b * 256 * 256;
        float* l2up_b  = l2up  + (size_t)b * 256 * 256;
        float* l1up_b  = l1up  + (size_t)b * 128 * 1024;
        float* l3_b    = outp + (size_t)b * 512;
        float* l0out_b = outp + (size_t)Bv * 512 + (size_t)b * 128 * Nv;

        // ---- SA1: N=8192 -> 1024 points, nsample 64, radius 0.2 ----
        fps_kernel<<<1, 256, 0, st>>>(pc, Nv, 1024, fpsidx);
        gather_rows3<<<G1(1024)>>>(pc, Nv, fpsidx, 1024, l1xyz_b);
        ball_query_kernel<<<G1(1024)>>>(pc, Nv, l1xyz_b, 1024, 64, 0.04f, ballidx);
        group_sa_kernel<<<G1(1024 * 64)>>>(pc, Nv, pc, 7, l1xyz_b, 1024, ballidx, 64, S0);
        run_conv(st, sa1c[0], S0, S1, 65536, true);
        run_conv(st, sa1c[1], S1, S2, 65536, true);
        run_conv(st, sa1c[2], S2, S0, 65536, true);
        maxpool_ns<<<G1(128 * 1024)>>>(S0, 128 * 1024, 64, l1_b);
        run_transformer(st, t1, l1_b, l1xyz_b, 128, 1024, knnidx,
                        Xs, Q, Kt, V, Prel, Kg, PE, T0, S1);

        // ---- SA2: 1024 -> 256 points, nsample 32, radius 0.4 ----
        fps_kernel<<<1, 256, 0, st>>>(l1xyz_b, 1024, 256, fpsidx);
        gather_rows3<<<G1(256)>>>(l1xyz_b, 1024, fpsidx, 256, l2xyz_b);
        ball_query_kernel<<<G1(256)>>>(l1xyz_b, 1024, l2xyz_b, 256, 32, 0.16f, ballidx);
        group_sa_kernel<<<G1(256 * 32)>>>(l1xyz_b, 1024, l1_b, 128, l2xyz_b, 256, ballidx, 32, S0);
        run_conv(st, sa2c[0], S0, S1, 8192, true);
        run_conv(st, sa2c[1], S1, S2, 8192, true);
        run_conv(st, sa2c[2], S2, S0, 8192, true);
        maxpool_ns<<<G1(256 * 256)>>>(S0, 256 * 256, 32, l2_b);
        run_transformer(st, t2, l2_b, l2xyz_b, 256, 256, knnidx,
                        Xs, Q, Kt, V, Prel, Kg, PE, T0, S1);

        // ---- SA3 (group-all): grouped = concat[l2_xyz, l2] (259 x 256) ----
        copy_flat<<<G1(3 * 256)>>>(l2xyz_b, S0, 3 * 256);
        copy_flat<<<G1(256 * 256)>>>(l2_b, S0 + 3 * 256, (size_t)256 * 256);
        run_conv(st, sa3c[0], S0, S1, 256, true);
        run_conv(st, sa3c[1], S1, S2, 256, true);
        run_conv(st, sa3c[2], S2, S1, 256, true);
        maxpool_ns<<<G1(512)>>>(S1, 512, 256, l3_b);           // l3 -> d_out

        // ---- FP3: interp(l3 @ l3_xyz=0) ++ l2 -> mlp -> l2_up ----
        three_interp<<<G1(256)>>>(l2xyz_b, 256, zero3, 1, l3_b, 512, S0);
        copy_flat<<<G1(256 * 256)>>>(l2_b, S0 + (size_t)512 * 256, (size_t)256 * 256);
        run_conv(st, fp3c[0], S0, S1, 256, true);
        run_conv(st, fp3c[1], S1, l2up_b, 256, true);
        run_conv(st, down2, l2_b, S2, 256, true);
        add_inplace<<<G1(256 * 256)>>>(l2up_b, S2, (size_t)256 * 256);
        run_transformer(st, te2, l2up_b, l2xyz_b, 256, 256, knnidx,
                        Xs, Q, Kt, V, Prel, Kg, PE, T0, S1);

        // ---- FP2: interp(l2_up @ l2_xyz) ++ l1 -> mlp -> l1_up ----
        three_interp<<<G1(1024)>>>(l1xyz_b, 1024, l2xyz_b, 256, l2up_b, 256, S0);
        copy_flat<<<G1(128 * 1024)>>>(l1_b, S0 + (size_t)256 * 1024, (size_t)128 * 1024);
        run_conv(st, fp2c[0], S0, S1, 1024, true);
        run_conv(st, fp2c[1], S1, l1up_b, 1024, true);
        run_conv(st, down1, l1_b, S2, 1024, true);
        add_inplace<<<G1(128 * 1024)>>>(l1up_b, S2, (size_t)128 * 1024);
        run_transformer(st, te1, l1up_b, l1xyz_b, 128, 1024, knnidx,
                        Xs, Q, Kt, V, Prel, Kg, PE, T0, S1);

        // ---- FP1: interp(l1_up @ l1_xyz) ++ [xyz,xyz] -> mlp -> l0_out ----
        three_interp<<<G1(Nv)>>>(pc, Nv, l1xyz_b, 1024, l1up_b, 128, S0);
        copy_flat<<<G1(3 * Nv)>>>(pc, S0 + (size_t)128 * Nv, (size_t)3 * Nv);
        copy_flat<<<G1(3 * Nv)>>>(pc, S0 + (size_t)131 * Nv, (size_t)3 * Nv);
        run_conv(st, fp1c[0], S0, S1, Nv, true);
        run_conv(st, fp1c[1], S1, S2, Nv, true);
        run_conv(st, fp1c[2], S2, l0out_b, Nv, true);          // -> d_out
    }
}
